// Hallucination_3685081940039
// MI455X (gfx1250) — matrix operand order repaired and checked by reading, not compiled, not run on hardware
//
#include <hip/hip_runtime.h>
#include <math.h>

typedef __attribute__((ext_vector_type(16))) _Float16 v16h;
typedef __attribute__((ext_vector_type(8)))  _Float16 v8h;
typedef __attribute__((ext_vector_type(8)))  float    v8f;

#define EPSF 1e-6f

// ---------------------------------------------------------------------------
// Frame construction matching reference _frames(): R columns = (e1,e2,e3), t=ca
// a9 = 3 atoms (n, ca, c) x 3 coords, contiguous.
// ---------------------------------------------------------------------------
__device__ __forceinline__ void make_frame(const float* __restrict__ a9,
                                           float R[3][3], float t[3]) {
    float n0 = a9[0], n1 = a9[1], n2 = a9[2];
    float c0 = a9[3], c1 = a9[4], c2 = a9[5];   // ca
    float C0 = a9[6], C1 = a9[7], C2 = a9[8];   // c
    float e10 = C0 - c0, e11 = C1 - c1, e12 = C2 - c2;
    float nrm = sqrtf(fmaxf(e10*e10 + e11*e11 + e12*e12, EPSF));
    e10 /= nrm; e11 /= nrm; e12 /= nrm;
    float v0 = n0 - c0, v1 = n1 - c1, v2 = n2 - c2;
    float dp = e10*v0 + e11*v1 + e12*v2;
    float f0 = v0 - dp*e10, f1 = v1 - dp*e11, f2 = v2 - dp*e12;
    float nrm2 = sqrtf(fmaxf(f0*f0 + f1*f1 + f2*f2, EPSF));
    f0 /= nrm2; f1 /= nrm2; f2 /= nrm2;
    float g0 = e11*f2 - e12*f1;
    float g1 = e12*f0 - e10*f2;
    float g2 = e10*f1 - e11*f0;
    R[0][0] = e10; R[1][0] = e11; R[2][0] = e12;
    R[0][1] = f0;  R[1][1] = f1;  R[2][1] = f2;
    R[0][2] = g0;  R[1][2] = g1;  R[2][2] = g2;
    t[0] = c0; t[1] = c1; t[2] = c2;
}

// ---------------------------------------------------------------------------
// Prep: padded atom mask (Nx16), res_mask, per-(l,i) frame feature vectors V.
// v = [1, ||t||^2+||s||^2-2 t'Ms, -2t+2Ms, -2s+2M't, -2M(9)], M = Rp*Rg^T
// ---------------------------------------------------------------------------
__global__ void prep_kernel(const float* __restrict__ pred,
                            const float* __restrict__ gt,
                            const float* __restrict__ am,
                            _Float16* __restrict__ Vbuf,
                            float* __restrict__ amP,
                            float* __restrict__ res_mask,
                            int L, int N, int A) {
    int tid = blockIdx.x * blockDim.x + threadIdx.x;
    if (tid < N * 16) {
        int i = tid >> 4, a = tid & 15;
        amP[tid] = (a < A) ? am[i * A + a] : 0.0f;
    }
    if (tid < N) {
        float any = 0.0f;
        for (int a = 0; a < A; ++a) any = fmaxf(any, am[tid * A + a]);
        res_mask[tid] = (any > 0.0f) ? 1.0f : 0.0f;
    }
    if (tid < L * N) {
        int l = tid / N, i = tid - l * N;
        float Rp[3][3], Rg[3][3], t[3], s[3];
        make_frame(&pred[(((size_t)l * N + i) * A) * 3], Rp, t);
        make_frame(&gt[((size_t)i * A) * 3], Rg, s);
        float M[3][3];
        #pragma unroll
        for (int d = 0; d < 3; ++d)
            #pragma unroll
            for (int e = 0; e < 3; ++e)
                M[d][e] = Rp[d][0]*Rg[e][0] + Rp[d][1]*Rg[e][1] + Rp[d][2]*Rg[e][2];
        float Ms[3], Mt[3];
        #pragma unroll
        for (int d = 0; d < 3; ++d)
            Ms[d] = M[d][0]*s[0] + M[d][1]*s[1] + M[d][2]*s[2];
        #pragma unroll
        for (int e = 0; e < 3; ++e)
            Mt[e] = M[0][e]*t[0] + M[1][e]*t[1] + M[2][e]*t[2];
        float tMs = t[0]*Ms[0] + t[1]*Ms[1] + t[2]*Ms[2];
        float v[17];
        v[0] = 1.0f;
        v[1] = (t[0]*t[0]+t[1]*t[1]+t[2]*t[2]) + (s[0]*s[0]+s[1]*s[1]+s[2]*s[2]) - 2.0f*tMs;
        #pragma unroll
        for (int d = 0; d < 3; ++d) v[2 + d] = -2.0f*t[d] + 2.0f*Ms[d];
        #pragma unroll
        for (int e = 0; e < 3; ++e) v[5 + e] = -2.0f*s[e] + 2.0f*Mt[e];
        #pragma unroll
        for (int d = 0; d < 3; ++d)
            #pragma unroll
            for (int e = 0; e < 3; ++e)
                v[8 + 3*d + e] = -2.0f * M[d][e];
        _Float16* vp = Vbuf + (size_t)tid * 32;
        #pragma unroll
        for (int k = 0; k < 32; ++k)
            vp[k] = (k < 17) ? (_Float16)v[k] : (_Float16)0.0f;
    }
}

// ---------------------------------------------------------------------------
// U build: per (l, j, a<16) feature vector u = [||p||^2+||g||^2, 1, p, g, p(x)g]
// ---------------------------------------------------------------------------
__global__ void u_kernel(const float* __restrict__ pred,
                         const float* __restrict__ gt,
                         _Float16* __restrict__ Ubuf,
                         int L, int N, int A) {
    int tid = blockIdx.x * blockDim.x + threadIdx.x;
    if (tid >= L * N * 16) return;
    int l = tid / (N * 16);
    int rem = tid - l * N * 16;
    int j = rem >> 4, a = rem & 15;
    _Float16* up = Ubuf + (size_t)tid * 32;
    if (a < A) {
        const float* pp = &pred[(((size_t)l * N + j) * A + a) * 3];
        const float* gp = &gt[((size_t)j * A + a) * 3];
        float p0 = pp[0], p1 = pp[1], p2 = pp[2];
        float g0 = gp[0], g1 = gp[1], g2 = gp[2];
        float u[17];
        u[0] = p0*p0 + p1*p1 + p2*p2 + g0*g0 + g1*g1 + g2*g2;
        u[1] = 1.0f;
        u[2] = p0; u[3] = p1; u[4] = p2;
        u[5] = g0; u[6] = g1; u[7] = g2;
        float pv[3] = {p0, p1, p2}, gv[3] = {g0, g1, g2};
        #pragma unroll
        for (int d = 0; d < 3; ++d)
            #pragma unroll
            for (int e = 0; e < 3; ++e)
                u[8 + 3*d + e] = pv[d] * gv[e];
        #pragma unroll
        for (int k = 0; k < 32; ++k)
            up[k] = (k < 17) ? (_Float16)u[k] : (_Float16)0.0f;
    } else {
        #pragma unroll
        for (int k = 0; k < 32; ++k) up[k] = (_Float16)0.0f;
    }
}

// ---------------------------------------------------------------------------
// Main WMMA kernel: one wave per (l, i_tile, j_block of 4).
//   A = U-tile (16 atoms of residue j x K=32)   -> rows M = atom a
//   B = V-tile (K=32 x 16 frames)               -> cols N = frame i (reused 4x)
//   D[a][i]: lane col = i_local, VGPR r -> a = r + 8h.
// Atom reduction = per-lane serial sum over 8 VGPRs + ONE cross-half shuffle.
// Low half lanes store one coalesced row of fapeT[l][j][.].
// ---------------------------------------------------------------------------
__global__ __launch_bounds__(256) void fape_wmma_kernel(
        const _Float16* __restrict__ Vbuf, const _Float16* __restrict__ Ubuf,
        const float* __restrict__ amP, const int* __restrict__ batch,
        float* __restrict__ fapeT, float* __restrict__ denT,
        float* __restrict__ out_final, float* __restrict__ out_roterr,
        int L, int N) {
    const int JB   = 4;
    const int lane = threadIdx.x & 31;
    const int wave = blockIdx.x * (blockDim.x >> 5) + (threadIdx.x >> 5);
    const int tilesI  = N >> 4;
    const int jBlocks = N / JB;
    const int wavesPerL = tilesI * jBlocks;
    const int l = wave / wavesPerL;
    if (l >= L) return;
    const int rem = wave - l * wavesPerL;
    const int it  = rem / jBlocks;           // i-tile
    const int jb  = (rem - it * jBlocks) * JB;
    const int h   = lane >> 4;               // half-wave group
    const int col = lane & 15;               // frame i_local (B col / D col)
    const int i   = it * 16 + col;           // this lane's frame index

    // B operand (V tile), constant across the 4 j's: element e -> K = 16h + e
    const _Float16* vp = Vbuf + ((size_t)(l * N + i)) * 32 + 16 * h;
    union { v16h v; v8h p[2]; } Bop;
    Bop.p[0] = *(const v8h*)(vp);
    Bop.p[1] = *(const v8h*)(vp + 8);

    // per-lane frame mask data: amP[i*16 + (8h + r)], r = 0..7  (two b128 loads)
    const float4 amI0 = *(const float4*)(amP + i * 16 + 8 * h);
    const float4 amI1 = *(const float4*)(amP + i * 16 + 8 * h + 4);
    const float  amIr[8] = {amI0.x, amI0.y, amI0.z, amI0.w,
                            amI1.x, amI1.y, amI1.z, amI1.w};
    const int bi = batch[i];

    for (int jj = 0; jj < JB; ++jj) {
        const int j = jb + jj;
        // A operand (U tile for residue j): row M = atom = col
        const _Float16* up = Ubuf + (((size_t)(l * N + j) * 16 + col)) * 32;
        union { v16h v; v8h p[2]; } Aop;
        Aop.p[0] = *(const v8h*)(up + 16 * h);
        Aop.p[1] = *(const v8h*)(up + 16 * h + 8);
        if (jj + 1 < JB) {  // prefetch next residue's U rows into L0
            __builtin_prefetch(up + 16 * 32, 0, 0);
        }

        v8f c = {};
        v8f d = __builtin_amdgcn_wmma_f32_16x16x32_f16(
            false, Aop.v, false, Bop.v, (short)0, c, false, false);

        // residue-j atom masks for a = 8h + r  (wave-uniform)
        const float4 amJ0 = *(const float4*)(amP + j * 16 + 8 * h);
        const float4 amJ1 = *(const float4*)(amP + j * 16 + 8 * h + 4);
        const float  amJr[8] = {amJ0.x, amJ0.y, amJ0.z, amJ0.w,
                                amJ1.x, amJ1.y, amJ1.z, amJ1.w};
        const float same = (batch[j] == bi) ? 1.0f : 0.0f;

        float num = 0.0f, den = 0.0f;
        #pragma unroll
        for (int r = 0; r < 8; ++r) {
            float m   = amIr[r] * amJr[r] * same;
            float err = __builtin_amdgcn_sqrtf(fmaxf(d[r], EPSF));
            num += m * err;
            den += m;
        }
        // fold the two atom halves (a=0..7 with a=8..15): one xor-16 shuffle each
        num += __shfl_xor(num, 16, 32);
        den += __shfl_xor(den, 16, 32);

        if (h == 0) {
            float f = num * __builtin_amdgcn_rcpf(fmaxf(den, EPSF));
            fapeT[((size_t)l * N + j) * N + i] = f;          // coalesced row
            if (l == 0) denT[(size_t)j * N + i] = den;       // coalesced row
            if (l == L - 1) {
                out_final[(size_t)i * N + j] = f;            // [i][j] scatter
                if (i == j) out_roterr[i] = f;
            }
        }
    }
}

// ---------------------------------------------------------------------------
// Chain-masked clipped sums over fapeT (transposed storage; all terms are
// symmetric in (i,j) so index roles are interchangeable).
// acc[0..1]=intraN[l], acc[2..3]=interN[l], acc[4]=same cnt, acc[5]=other cnt.
// ---------------------------------------------------------------------------
__global__ void chain_reduce(const float* __restrict__ fapeT,
                             const float* __restrict__ denT,
                             const int* __restrict__ chain,
                             float* __restrict__ acc, int L, int N) {
    __shared__ float sm[256];
    size_t total = (size_t)L * N * N;
    float loc[6] = {0.f, 0.f, 0.f, 0.f, 0.f, 0.f};
    for (size_t idx = (size_t)blockIdx.x * blockDim.x + threadIdx.x;
         idx < total; idx += (size_t)gridDim.x * blockDim.x) {
        int l  = (int)(idx / ((size_t)N * N));
        int rr = (int)(idx - (size_t)l * N * N);
        int a = rr / N, b = rr - (rr / N) * N;
        float dn  = denT[rr];
        float pm2 = (dn > 0.f) ? 1.f : 0.f;
        float sc  = (chain[a] == chain[b]) ? pm2 : 0.f;
        float oc  = pm2 - sc;
        float cl  = fminf(fmaxf(fapeT[idx], 0.f), 10.f);
        loc[l]     += cl * sc;
        loc[2 + l] += cl * oc;
        if (l == 0) { loc[4] += sc; loc[5] += oc; }
    }
    for (int k = 0; k < 6; ++k) {
        sm[threadIdx.x] = loc[k];
        __syncthreads();
        for (int s = 128; s > 0; s >>= 1) {
            if ((int)threadIdx.x < s) sm[threadIdx.x] += sm[threadIdx.x + s];
            __syncthreads();
        }
        if (threadIdx.x == 0) atomicAdd(&acc[k], sm[0]);
        __syncthreads();
    }
}

// ---------------------------------------------------------------------------
// lDDT: one thread per residue i, loop over j.
// ---------------------------------------------------------------------------
__global__ void lddt_kernel(const float* __restrict__ pred,
                            const float* __restrict__ gt,
                            const float* __restrict__ res_mask,
                            float* __restrict__ out_lddt,
                            int L, int N, int A) {
    int i = blockIdx.x * blockDim.x + threadIdx.x;
    if (i >= N) return;
    const float* gi = &gt[((size_t)i * A + 1) * 3];
    const float* pi = &pred[(((size_t)(L - 1) * N + i) * A + 1) * 3];
    float gx = gi[0], gy = gi[1], gz = gi[2];
    float px = pi[0], py = pi[1], pz = pi[2];
    float rmi = res_mask[i];
    float s = 0.f, cnt = 0.f;
    for (int j = 0; j < N; ++j) {
        if (j == i) continue;
        const float* gj = &gt[((size_t)j * A + 1) * 3];
        float dx = gx - gj[0], dy = gy - gj[1], dz = gz - gj[2];
        float gd = __builtin_amdgcn_sqrtf(fmaxf(dx*dx + dy*dy + dz*dz, EPSF));
        if (gd < 15.0f && rmi > 0.f && res_mask[j] > 0.f) {
            const float* pj = &pred[(((size_t)(L - 1) * N + j) * A + 1) * 3];
            float ex = px - pj[0], ey = py - pj[1], ez = pz - pj[2];
            float pd = __builtin_amdgcn_sqrtf(fmaxf(ex*ex + ey*ey + ez*ez, EPSF));
            float de = fabsf(gd - pd);
            float lv = ((de < 0.5f ? 1.f : 0.f) + (de < 1.0f ? 1.f : 0.f) +
                        (de < 2.0f ? 1.f : 0.f) + (de < 4.0f ? 1.f : 0.f)) * 0.25f;
            s += lv; cnt += 1.f;
        }
    }
    out_lddt[i] = s / fmaxf(cnt, EPSF);
}

// ---------------------------------------------------------------------------
// Finalize: deterministic serial diag / res_mask sums + scalar outputs.
// out[0..L-1]=fape_loss, out[L..2L-1]=rotamer_loss   (diag of fapeT == diag)
// ---------------------------------------------------------------------------
__global__ void finalize_kernel(const float* __restrict__ acc,
                                const float* __restrict__ fapeT,
                                const float* __restrict__ res_mask,
                                float* __restrict__ out, int L, int N) {
    if (threadIdx.x != 0 || blockIdx.x != 0) return;
    float nres = 0.f;
    for (int i = 0; i < N; ++i) nres += res_mask[i];
    float scC = fmaxf(acc[4], EPSF), ocC = fmaxf(acc[5], EPSF);
    for (int l = 0; l < L; ++l) {
        out[l] = acc[l] / scC / 10.0f + acc[2 + l] / ocC / 10.0f;
        float rs = 0.f;
        for (int i = 0; i < N; ++i) rs += fapeT[((size_t)l * N + i) * N + i];
        out[L + l] = rs / fmaxf(nres, EPSF);
    }
}

// ---------------------------------------------------------------------------
extern "C" void kernel_launch(void* const* d_in, const int* in_sizes, int n_in,
                              void* d_out, int out_size, void* d_ws, size_t ws_size,
                              hipStream_t stream) {
    const float* pred = (const float*)d_in[0];
    const float* gt   = (const float*)d_in[1];
    const float* am   = (const float*)d_in[2];
    const int* batch  = (const int*)d_in[3];
    const int* chain  = (const int*)d_in[4];
    const int N = in_sizes[3];
    const int A = in_sizes[2] / N;
    const int L = in_sizes[0] / (N * A * 3);
    float* out = (float*)d_out;

    // workspace carve-out (256B aligned blocks)
    char* ws = (char*)d_ws;
    size_t off = 0;
    auto carve = [&](size_t bytes) -> void* {
        void* p = ws + off;
        off += (bytes + 255) & ~(size_t)255;
        return p;
    };
    float*     acc      = (float*)carve(16 * sizeof(float));
    float*     amP      = (float*)carve((size_t)N * 16 * sizeof(float));
    float*     res_mask = (float*)carve((size_t)N * sizeof(float));
    _Float16*  Vbuf     = (_Float16*)carve((size_t)L * N * 32 * sizeof(_Float16));
    _Float16*  Ubuf     = (_Float16*)carve((size_t)L * N * 16 * 32 * sizeof(_Float16));
    float*     fapeT    = (float*)carve((size_t)L * N * N * sizeof(float));
    float*     denT     = (float*)carve((size_t)N * N * sizeof(float));

    float* out_fape_loss = out;                          // L
    float* out_final     = out + 2 * L;                  // N*N  ([i][j])
    float* out_roterrD   = out + 2 * L + (size_t)N * N;  // N
    float* out_lddt      = out_roterrD + N;              // N

    hipMemsetAsync(acc, 0, 16 * sizeof(float), stream);

    {
        int total = (N * 16 > L * N) ? N * 16 : L * N;
        prep_kernel<<<(total + 255) / 256, 256, 0, stream>>>(
            pred, gt, am, Vbuf, amP, res_mask, L, N, A);
    }
    {
        int total = L * N * 16;
        u_kernel<<<(total + 255) / 256, 256, 0, stream>>>(pred, gt, Ubuf, L, N, A);
    }
    {
        int waves = L * (N / 16) * (N / 4);
        fape_wmma_kernel<<<(waves + 7) / 8, 256, 0, stream>>>(
            Vbuf, Ubuf, amP, batch, fapeT, denT, out_final, out_roterrD, L, N);
    }
    chain_reduce<<<512, 256, 0, stream>>>(fapeT, denT, chain, acc, L, N);
    lddt_kernel<<<(N + 255) / 256, 256, 0, stream>>>(pred, gt, res_mask, out_lddt, L, N, A);
    finalize_kernel<<<1, 32, 0, stream>>>(acc, fapeT, res_mask, out_fape_loss, L, N);
}
